// MultiHeadAttention_4999341932927
// MI455X (gfx1250) — matrix operand order repaired and checked by reading, not compiled, not run on hardware
//
#include <hip/hip_runtime.h>
#include <hip/hip_bf16.h>

// ---------------------------------------------------------------------------
// MultiHeadAttention for gfx1250 (MI455X).
// All matmuls on v_wmma_f32_16x16x32_bf16. fp32->bf16 conversion hoisted into
// one-shot pre-pass kernels. Tile staging via async global->LDS (ASYNCcnt),
// double-buffered with partial s_wait_asynccnt waits to hide HBM latency.
// ---------------------------------------------------------------------------

typedef __attribute__((ext_vector_type(16))) __bf16 v16bf;
typedef __attribute__((ext_vector_type(8)))  float  v8f;
typedef __attribute__((ext_vector_type(4)))  int    vi4;

union BF16Frag { v16bf v; unsigned u[8]; };
union F32Acc   { v8f  v; float    f[8]; };

static __device__ __forceinline__ unsigned short f2bf(float f) {
    union { float f; unsigned u; } x; x.f = f;
    unsigned r = x.u + 0x7FFFu + ((x.u >> 16) & 1u);   // round-to-nearest-even
    return (unsigned short)(r >> 16);
}

static __device__ __forceinline__ v8f wmma_bf16(v16bf a, v16bf b, v8f c) {
    return __builtin_amdgcn_wmma_f32_16x16x32_bf16(
        false, a, false, b, (short)0, c, false, false);
}

// A-fragment (16x32 bf16) per-lane K offset for VGPR slot v.
static __device__ __forceinline__ int a_kk(int v, int laneHi) {
    return (v < 4 ? 2 * v : 16 + 2 * (v - 4)) + (laneHi ? 8 : 0);
}

// ---------------- async global->LDS staging (CDNA5 ASYNCcnt path) ----------
#if defined(__HIP_DEVICE_COMPILE__) && __has_builtin(__builtin_amdgcn_global_load_async_to_lds_b128)
#define HAVE_ASYNC_LDS 1
#else
#define HAVE_ASYNC_LDS 0
#endif

#if HAVE_ASYNC_LDS
typedef __attribute__((address_space(1))) vi4 g_vi4;   // global int4
typedef __attribute__((address_space(3))) vi4 l_vi4;   // LDS int4
static __device__ __forceinline__ void stage16(const void* g, void* l) {
    __builtin_amdgcn_global_load_async_to_lds_b128((g_vi4*)g, (l_vi4*)l, 0, 0);
}
#if __has_builtin(__builtin_amdgcn_s_wait_asynccnt)
#define WAIT_ASYNC(n) __builtin_amdgcn_s_wait_asynccnt(n)
#else
#define WAIT_ASYNC(n) asm volatile("s_wait_asynccnt %0" :: "i"(n) : "memory")
#endif
#else
static __device__ __forceinline__ void stage16(const void* g, void* l) {
    *(uint4*)l = *(const uint4*)g;
}
#define WAIT_ASYNC(n) ((void)0)
#endif

// ---------------------------------------------------------------------------
// Pre-pass 1: elementwise fp32 -> bf16 (4 elems/thread).
// ---------------------------------------------------------------------------
__global__ __launch_bounds__(256) void cvt_bf16_kernel(
    const float* __restrict__ src, unsigned short* __restrict__ dst)
{
    size_t i = ((size_t)blockIdx.x * 256 + threadIdx.x) * 4;
    float4 f = *(const float4*)(src + i);
    ushort4 o;
    o.x = f2bf(f.x); o.y = f2bf(f.y); o.z = f2bf(f.z); o.w = f2bf(f.w);
    *(ushort4*)(dst + i) = o;
}

// ---------------------------------------------------------------------------
// Pre-pass 2: W[K][N] fp32 -> Wt[N][K] bf16 (32x32 LDS tile transpose).
// Grid (N/32, K/32), block (32,8).
// ---------------------------------------------------------------------------
__global__ __launch_bounds__(256) void transpose_cvt_kernel(
    const float* __restrict__ W, unsigned short* __restrict__ Wt, int K, int N)
{
    __shared__ unsigned short tile[32][33];
    int nt = blockIdx.x * 32, kt = blockIdx.y * 32;
    #pragma unroll
    for (int i = 0; i < 4; ++i) {
        int r = threadIdx.y + i * 8;
        tile[r][threadIdx.x] = f2bf(W[(size_t)(kt + r) * N + nt + threadIdx.x]);
    }
    __syncthreads();
    #pragma unroll
    for (int i = 0; i < 4; ++i) {
        int r = threadIdx.y + i * 8;
        Wt[(size_t)(nt + r) * K + kt + threadIdx.x] = tile[threadIdx.x][r];
    }
}

// ---------------------------------------------------------------------------
// Pre-pass 3: Vh[b*S+s][h*64+dv] bf16 -> VhT[(b*16+h)*64+dv][s] bf16.
// Grid (S/32, 64/32, B*H), block (32,8).
// ---------------------------------------------------------------------------
__global__ __launch_bounds__(256) void transpose_vh_kernel(
    const unsigned short* __restrict__ Vh, unsigned short* __restrict__ VhT)
{
    __shared__ unsigned short tile[32][33];
    const int bh = blockIdx.z;            // b*16+h
    const int b = bh >> 4, h = bh & 15;
    const int st = blockIdx.x * 32, dt = blockIdx.y * 32;
    #pragma unroll
    for (int i = 0; i < 4; ++i) {
        int s = threadIdx.y + i * 8;
        tile[s][threadIdx.x] =
            Vh[(size_t)(b * 2048 + st + s) * 1024 + h * 64 + dt + threadIdx.x];
    }
    __syncthreads();
    #pragma unroll
    for (int i = 0; i < 4; ++i) {
        int d = threadIdx.y + i * 8;
        VhT[(size_t)(bh * 64 + dt + d) * 2048 + st + threadIdx.x] = tile[threadIdx.x][d];
    }
}

// ---------------------------------------------------------------------------
// GEMM: C[M,N] = A[M,K](bf16) @ Wt[N][K](bf16, pre-transposed) + bias, scaled.
// Block 256 (8 waves), tile 64(M) x 128(N), K-step 32, double-buffered LDS.
// Per wave: 16x64 output, 4 WMMAs per K-step.
// ---------------------------------------------------------------------------
template <bool OUT_BF16>
__global__ __launch_bounds__(256) void gemm_bias_kernel(
    const unsigned short* __restrict__ A, const unsigned short* __restrict__ Wt,
    const float* __restrict__ bias, void* __restrict__ Cptr,
    int M, int K, int N, float scale)
{
    __shared__ unsigned short Alds[2][64][40];    // 80B row stride (16B aligned)
    __shared__ unsigned short Btlds[2][128][40];  // [n][k]

    const int tid    = threadIdx.x;
    const int lane   = tid & 31;
    const int wave   = tid >> 5;
    const int laneN  = lane & 15;
    const int laneHi = lane >> 4;

    const int mBlock = blockIdx.x * 64;
    const int nBlock = blockIdx.y * 128;
    const int wm = wave >> 1;
    const int wn = wave & 1;

    F32Acc acc[4];
    #pragma unroll
    for (int i = 0; i < 4; ++i)
        #pragma unroll
        for (int e = 0; e < 8; ++e) acc[i].f[e] = 0.0f;

    // per-thread staging coordinates (16B granules)
    const int ar = tid >> 2, ac = (tid & 3) * 8;        // A: 64x32, 1x16B/thread
    const int bn = tid >> 1, bc = (tid & 1) * 16;       // B: 128x32, 2x16B/thread
    const unsigned short* gA = A  + (size_t)(mBlock + ar) * K + ac;
    const unsigned short* gB = Wt + (size_t)(nBlock + bn) * K + bc;

    // prologue: stage k0=0 into buffer 0 (3 async ops / thread)
    stage16(gA,     &Alds[0][ar][ac]);
    stage16(gB,     &Btlds[0][bn][bc]);
    stage16(gB + 8, &Btlds[0][bn][bc + 8]);

    for (int k0 = 0; k0 < K; k0 += 32) {
        const int buf = (k0 >> 5) & 1;
        const int k1  = k0 + 32;
        if (k1 < K) {   // stage next tile into the other buffer
            stage16(gA + k1,     &Alds[buf ^ 1][ar][ac]);
            stage16(gB + k1,     &Btlds[buf ^ 1][bn][bc]);
            stage16(gB + k1 + 8, &Btlds[buf ^ 1][bn][bc + 8]);
            WAIT_ASYNC(3);      // current buffer's (older) loads complete
        } else {
            WAIT_ASYNC(0);
        }
        __syncthreads();

        BF16Frag a;
        #pragma unroll
        for (int v = 0; v < 8; ++v)
            a.u[v] = *(const unsigned*)&Alds[buf][wm * 16 + laneN][a_kk(v, laneHi)];

        #pragma unroll
        for (int nt = 0; nt < 4; ++nt) {
            BF16Frag b;
            #pragma unroll
            for (int v = 0; v < 8; ++v)
                b.u[v] = *(const unsigned*)&Btlds[buf][wn * 64 + nt * 16 + laneN]
                                                 [a_kk(v, laneHi)];
            acc[nt].v = wmma_bf16(a.v, b.v, acc[nt].v);
        }
        __syncthreads();   // before next iteration overwrites buf^1
    }

    #pragma unroll
    for (int nt = 0; nt < 4; ++nt) {
        int col = nBlock + wn * 64 + nt * 16 + laneN;
        float bval = bias[col];
        #pragma unroll
        for (int e = 0; e < 8; ++e) {
            int row = mBlock + wm * 16 + e + laneHi * 8;
            float r = (acc[nt].f[e] + bval) * scale;
            if (OUT_BF16)
                ((unsigned short*)Cptr)[(size_t)row * N + col] = f2bf(r);
            else
                ((float*)Cptr)[(size_t)row * N + col] = r;
        }
    }
}

// ---------------------------------------------------------------------------
// Flash attention: grid (S/64, H, B), block 128 (4 waves).
// Wave owns a 16-query tile; 32-key chunks, double-buffered staging.
// K natural layout, V pre-transposed. Qh pre-scaled by 1/sqrt(dk).
// ---------------------------------------------------------------------------
__global__ __launch_bounds__(128) void attn_kernel(
    const unsigned short* __restrict__ Qh, const unsigned short* __restrict__ Kh,
    const unsigned short* __restrict__ VhT, unsigned short* __restrict__ Out)
{
    constexpr int S  = 2048;
    constexpr int HD = 1024;

    __shared__ unsigned short Klds[2][32][64];    // [key][dk]  (128B rows)
    __shared__ unsigned short Vtlds[2][64][32];   // [dv][key]  (64B rows)
    __shared__ unsigned short Plds[4][16][36];    // per-wave P scratch [m][key]

    const int tid    = threadIdx.x;
    const int lane   = tid & 31;
    const int wave   = tid >> 5;
    const int laneN  = lane & 15;
    const int laneHi = lane >> 4;

    const int h  = blockIdx.y;
    const int b  = blockIdx.z;
    const int q0 = blockIdx.x * 64 + wave * 16;

    // Q fragments (dk=64 -> two 16x32 A-frags) held in VGPRs for the whole loop
    BF16Frag qf[2];
    #pragma unroll
    for (int i = 0; i < 2; ++i)
        #pragma unroll
        for (int v = 0; v < 8; ++v) {
            size_t idx = (size_t)(b * S + q0 + laneN) * HD
                       + h * 64 + i * 32 + a_kk(v, laneHi);
            qf[i].u[v] = *(const unsigned*)(Qh + idx);
        }

    F32Acc o[4];
    #pragma unroll
    for (int nt = 0; nt < 4; ++nt)
        #pragma unroll
        for (int e = 0; e < 8; ++e) o[nt].f[e] = 0.0f;

    float mrun[8], lrun[8];
    #pragma unroll
    for (int e = 0; e < 8; ++e) { mrun[e] = -3.0e38f; lrun[e] = 0.0f; }

    // staging coordinates: K chunk 32x64 (4KB), Vt chunk 64x32 (4KB); 32B/thread
    const int kKey = tid >> 2, kCol = (tid & 3) * 16;   // elems
    const int vDv  = tid >> 1, vOff = (tid & 1) * 16;   // elems
    const unsigned short* gK = Kh + (size_t)(b * S + kKey) * HD + h * 64 + kCol;
    const unsigned short* gV = VhT + ((size_t)((b * 16 + h) * 64) + vDv) * S + vOff;

    // prologue: stage chunk kc=0 into buffer 0 (4 async ops / thread)
    stage16(gK,     &Klds[0][kKey][kCol]);
    stage16(gK + 8, &Klds[0][kKey][kCol + 8]);
    stage16(gV,     &Vtlds[0][vDv][vOff]);
    stage16(gV + 8, &Vtlds[0][vDv][vOff + 8]);

    for (int kc = 0; kc < S; kc += 32) {
        const int buf = (kc >> 5) & 1;
        const int kn  = kc + 32;
        if (kn < S) {   // stage next chunk into the other buffer
            stage16(gK + (size_t)kn * HD,     &Klds[buf ^ 1][kKey][kCol]);
            stage16(gK + (size_t)kn * HD + 8, &Klds[buf ^ 1][kKey][kCol + 8]);
            stage16(gV + kn,                  &Vtlds[buf ^ 1][vDv][vOff]);
            stage16(gV + kn + 8,              &Vtlds[buf ^ 1][vDv][vOff + 8]);
            WAIT_ASYNC(4);
        } else {
            WAIT_ASYNC(0);
        }
        __syncthreads();

        // scores: two 16x16 C-frags; K tile rows are exactly B^T layout
        F32Acc sc[2];
        #pragma unroll
        for (int sub = 0; sub < 2; ++sub) {
            #pragma unroll
            for (int e = 0; e < 8; ++e) sc[sub].f[e] = 0.0f;
            #pragma unroll
            for (int i = 0; i < 2; ++i) {
                BF16Frag bfr;
                #pragma unroll
                for (int v = 0; v < 8; ++v)
                    bfr.u[v] = *(const unsigned*)&Klds[buf][sub * 16 + laneN]
                                                      [i * 32 + a_kk(v, laneHi)];
                sc[sub].v = wmma_bf16(qf[i].v, bfr.v, sc[sub].v);
            }
        }

        // online softmax stats (16-lane half reductions)
        float alpha[8];
        #pragma unroll
        for (int e = 0; e < 8; ++e) {
            float mt = fmaxf(sc[0].f[e], sc[1].f[e]);
            #pragma unroll
            for (int mask = 1; mask <= 8; mask <<= 1)
                mt = fmaxf(mt, __shfl_xor(mt, mask, 32));
            float mnew = fmaxf(mrun[e], mt);
            alpha[e] = __expf(mrun[e] - mnew);
            mrun[e] = mnew;
            sc[0].f[e] = __expf(sc[0].f[e] - mnew);
            sc[1].f[e] = __expf(sc[1].f[e] - mnew);
            float rs = sc[0].f[e] + sc[1].f[e];
            #pragma unroll
            for (int mask = 1; mask <= 8; mask <<= 1)
                rs += __shfl_xor(rs, mask, 32);
            lrun[e] = lrun[e] * alpha[e] + rs;
        }

        #pragma unroll
        for (int nt = 0; nt < 4; ++nt)
            #pragma unroll
            for (int e = 0; e < 8; ++e) o[nt].f[e] *= alpha[e];

        // P: C-layout f32 -> LDS -> A-layout bf16 (per-wave scratch)
        #pragma unroll
        for (int sub = 0; sub < 2; ++sub)
            #pragma unroll
            for (int e = 0; e < 8; ++e)
                Plds[wave][e + laneHi * 8][sub * 16 + laneN] = f2bf(sc[sub].f[e]);

        BF16Frag pf;
        #pragma unroll
        for (int v = 0; v < 8; ++v)
            pf.u[v] = *(const unsigned*)&Plds[wave][laneN][a_kk(v, laneHi)];

        // O += P[16x32] x V[32x64]; Vt rows give aligned b32 pair loads
        #pragma unroll
        for (int nt = 0; nt < 4; ++nt) {
            BF16Frag vf;
            #pragma unroll
            for (int v = 0; v < 8; ++v)
                vf.u[v] = *(const unsigned*)&Vtlds[buf][nt * 16 + laneN]
                                                  [a_kk(v, laneHi)];
            o[nt].v = wmma_bf16(pf.v, vf.v, o[nt].v);
        }
        __syncthreads();   // before next iteration overwrites buf^1
    }

    #pragma unroll
    for (int nt = 0; nt < 4; ++nt)
        #pragma unroll
        for (int e = 0; e < 8; ++e) {
            int qi  = q0 + e + laneHi * 8;
            int col = h * 64 + nt * 16 + laneN;
            Out[(size_t)(b * S + qi) * HD + col] = f2bf(o[nt].f[e] / lrun[e]);
        }
}

// ---------------------------------------------------------------------------
extern "C" void kernel_launch(void* const* d_in, const int* in_sizes, int n_in,
                              void* d_out, int out_size, void* d_ws, size_t ws_size,
                              hipStream_t stream) {
    (void)in_sizes; (void)n_in; (void)out_size; (void)ws_size;

    const float* q  = (const float*)d_in[0];
    const float* k  = (const float*)d_in[1];
    const float* v  = (const float*)d_in[2];
    const float* Wq = (const float*)d_in[3];
    const float* bq = (const float*)d_in[4];
    const float* Wk = (const float*)d_in[5];
    const float* bk = (const float*)d_in[6];
    const float* Wv = (const float*)d_in[7];
    const float* bv = (const float*)d_in[8];
    const float* Wo = (const float*)d_in[9];
    const float* bo = (const float*)d_in[10];
    float* out = (float*)d_out;

    const int M = 4 * 2048, K = 1024, N = 1024;
    const size_t MB = 1024 * 1024;

    // workspace layout (bf16 buffers); VhT / Ao alias dead input slots
    unsigned short* WqT = (unsigned short*)((char*)d_ws + 0 * MB);   // 2 MB
    unsigned short* WkT = (unsigned short*)((char*)d_ws + 2 * MB);
    unsigned short* WvT = (unsigned short*)((char*)d_ws + 4 * MB);
    unsigned short* WoT = (unsigned short*)((char*)d_ws + 6 * MB);
    unsigned short* qb  = (unsigned short*)((char*)d_ws + 8 * MB);   // 16 MB each
    unsigned short* kb  = (unsigned short*)((char*)d_ws + 24 * MB);
    unsigned short* vb  = (unsigned short*)((char*)d_ws + 40 * MB);
    unsigned short* Qh  = (unsigned short*)((char*)d_ws + 56 * MB);
    unsigned short* Kh  = (unsigned short*)((char*)d_ws + 72 * MB);
    unsigned short* Vh  = (unsigned short*)((char*)d_ws + 88 * MB);
    unsigned short* VhT = qb;   // qb dead after projections
    unsigned short* Ao  = kb;   // kb dead after projections

    // 1) convert activations to bf16
    const int cvt_grid = (M * K) / (256 * 4);
    cvt_bf16_kernel<<<cvt_grid, 256, 0, stream>>>(q, qb);
    cvt_bf16_kernel<<<cvt_grid, 256, 0, stream>>>(k, kb);
    cvt_bf16_kernel<<<cvt_grid, 256, 0, stream>>>(v, vb);

    // 2) transpose-convert weights to Wt[N][K] bf16
    dim3 tb(32, 8), tg(N / 32, K / 32);
    transpose_cvt_kernel<<<tg, tb, 0, stream>>>(Wq, WqT, K, N);
    transpose_cvt_kernel<<<tg, tb, 0, stream>>>(Wk, WkT, K, N);
    transpose_cvt_kernel<<<tg, tb, 0, stream>>>(Wv, WvT, K, N);
    transpose_cvt_kernel<<<tg, tb, 0, stream>>>(Wo, WoT, K, N);

    // 3) QKV projections (Q pre-scaled by 1/sqrt(64))
    dim3 gemm_grid(M / 64, N / 128);
    gemm_bias_kernel<true><<<gemm_grid, 256, 0, stream>>>(qb, WqT, bq, Qh, M, K, N, 0.125f);
    gemm_bias_kernel<true><<<gemm_grid, 256, 0, stream>>>(kb, WkT, bk, Kh, M, K, N, 1.0f);
    gemm_bias_kernel<true><<<gemm_grid, 256, 0, stream>>>(vb, WvT, bv, Vh, M, K, N, 1.0f);

    // 4) transpose V per (b,h) for the PV matmul B-operand
    dim3 vtg(2048 / 32, 64 / 32, 4 * 16);
    transpose_vh_kernel<<<vtg, tb, 0, stream>>>(Vh, VhT);

    // 5) flash attention
    dim3 attn_grid(2048 / 64, 16, 4);
    attn_kernel<<<attn_grid, 128, 0, stream>>>(Qh, Kh, VhT, Ao);

    // 6) output projection -> fp32
    gemm_bias_kernel<false><<<gemm_grid, 256, 0, stream>>>(Ao, WoT, bo, out, M, K, N, 1.0f);
}
